// LambdaLayer_77257871721004
// MI455X (gfx1250) — compile-verified, matrix operand order checked by reading, not run on hardware
//
#include <hip/hip_runtime.h>

// ---------------------------------------------------------------------------
// LambdaLayer on MI455X (gfx1250): bf16 WMMA pipeline
//   K0: prep (BN fold, weight swizzle to WMMA frag layout)
//   K1: fused q/k/v projection GEMM (WMMA bf16) + BN epilogue
//   K2: softmax over positions for k
//   K3: Lc = softmax(k) @ v^T   (WMMA bf16, K=4096)
//   K4: positional conv (im2col GEMM on WMMA, taps = K dim) + combine with q
//       - LDS staged via GLOBAL_LOAD_ASYNC_TO_LDS (ASYNCcnt) when available
// ---------------------------------------------------------------------------

typedef __attribute__((ext_vector_type(16))) __bf16 v16bf;
typedef __attribute__((ext_vector_type(2)))  __bf16 v2bf;
typedef __attribute__((ext_vector_type(8)))  float  v8f;
typedef __attribute__((ext_vector_type(4)))  unsigned int u32x4;
typedef __attribute__((ext_vector_type(4)))  int    i32x4v;

union Frag16 {
    v16bf v;
    u32x4 q[2];
    unsigned int w[8];
    unsigned short u[16];
};

#define NB     16          // batch
#define NPOS   4096        // 64*64
#define CIN    256
#define NCOL   144         // 64 q + 16 k + 64 v
#define RAD    11
#define RT     23
#define TAPS   529         // 23*23
#define KTC    17          // ceil(544/32) K-tiles for conv (taps padded to 544)
#define PSTR   42          // halo patch row stride (elements); origin col is even

// ---- workspace layout (bytes) ----
static constexpr size_t WQKV_OFF  = 0;                                  // 8*9*32*16 bf16 = 73728
static constexpr size_t WSW_OFF   = WQKV_OFF + 8*9*32*16*2;             // 17*32*16 bf16 = 17408
static constexpr size_t SCALE_OFF = WSW_OFF + (size_t)KTC*32*16*2;      // 144 f32
static constexpr size_t BIAS_OFF  = SCALE_OFF + 144*4;                  // 144 f32
static constexpr size_t Q_OFF     = ((BIAS_OFF + 144*4 + 255)/256)*256; // 65536*64 f32
static constexpr size_t KRAW_OFF  = Q_OFF   + (size_t)NB*NPOS*64*4;     // 16*16*4096 f32
static constexpr size_t KSM_OFF   = KRAW_OFF+ (size_t)NB*16*NPOS*4;     // 16*16*4096 bf16
static constexpr size_t VBF_OFF   = KSM_OFF + (size_t)NB*16*NPOS*2;     // 16*64*4096 bf16
static constexpr size_t LC_OFF    = VBF_OFF + (size_t)NB*64*NPOS*2;     // 16*16*64 f32

// native fptrunc -> backend picks hardware cvt if available
__device__ __forceinline__ unsigned short f2bf(float f) {
    __bf16 h = (__bf16)f;
    return __builtin_bit_cast(unsigned short, h);
}

__device__ __forceinline__ unsigned int pk_bf16(float a, float b) {
#if __has_builtin(__builtin_amdgcn_cvt_pk_bf16_f32)
    v2bf r = __builtin_amdgcn_cvt_pk_bf16_f32(a, b);
    return __builtin_bit_cast(unsigned int, r);
#else
    v2bf r;
    r[0] = (__bf16)a;
    r[1] = (__bf16)b;
    return __builtin_bit_cast(unsigned int, r);
#endif
}

// ---------------------------------------------------------------------------
// K0: prep — swizzle weights into WMMA fragment order, fold BN into scale/bias
// frag element e (0..7 -> K=base..base+7, 8..15 -> K=base+16..base+23),
// base = (lane>=16)?8:0, lane&15 = column (B frag) / row (A frag).
// ---------------------------------------------------------------------------
__global__ __launch_bounds__(256) void k_prep(
    const float* __restrict__ Wq, const float* __restrict__ Wk, const float* __restrict__ Wv,
    const float* __restrict__ gq, const float* __restrict__ bq,
    const float* __restrict__ mq, const float* __restrict__ vq,
    const float* __restrict__ gv, const float* __restrict__ bv,
    const float* __restrict__ mv, const float* __restrict__ vv,
    const float* __restrict__ pw,
    unsigned short* __restrict__ wqkv_sw, unsigned short* __restrict__ wsw,
    float* __restrict__ scale, float* __restrict__ bias)
{
    int gid = blockIdx.x * blockDim.x + threadIdx.x;
    const int N1 = 8 * 9 * 32 * 16;         // projection weight frags
    const int N2 = KTC * 32 * 16;           // conv weight frags (A layout)
    if (gid < N1) {
        int idx = gid;
        int e  = idx & 15; idx >>= 4;
        int ln = idx & 31; idx >>= 5;
        int nt = idx % 9;
        int kt = idx / 9;
        int khalf = (ln >= 16) ? 8 : 0;
        int koff  = (e < 8) ? (khalf + e) : (khalf + 8 + e);
        int k     = kt * 32 + koff;
        int col   = nt * 16 + (ln & 15);
        float w;
        if      (col < 64) w = Wq[k * 64 + col];
        else if (col < 80) w = Wk[k * 16 + (col - 64)];
        else               w = Wv[k * 64 + (col - 80)];
        wqkv_sw[gid] = f2bf(w);
        return;
    }
    gid -= N1;
    if (gid < N2) {
        int idx = gid;
        int e  = idx & 15; idx >>= 4;
        int ln = idx & 31; idx >>= 5;
        int kt = idx;                       // 0..16
        int khalf = (ln >= 16) ? 8 : 0;
        int koff  = (e < 8) ? (khalf + e) : (khalf + 8 + e);
        int t     = kt * 32 + koff;         // tap index
        int kch   = ln & 15;                // output k-channel (A-matrix row)
        float w   = (t < TAPS) ? pw[t * 16 + kch] : 0.0f;
        wsw[gid] = f2bf(w);
        return;
    }
    gid -= N2;
    if (gid < NCOL) {
        float s, b;
        if (gid < 64) {
            s = gq[gid] * rsqrtf(vq[gid] + 1e-3f);
            b = bq[gid] - mq[gid] * s;
        } else if (gid < 80) {
            s = 1.0f; b = 0.0f;             // k channels: no BN
        } else {
            int c = gid - 80;
            s = gv[c] * rsqrtf(vv[c] + 1e-3f);
            b = bv[c] - mv[c] * s;
        }
        scale[gid] = s;
        bias[gid]  = b;
    }
}

// ---------------------------------------------------------------------------
// K1: projection GEMM  (65536 x 256) x (256 x 144), WMMA bf16, fused BN.
// 256 threads = 8 waves; wave computes a 16-row x 144-col strip.
// ---------------------------------------------------------------------------
__global__ __launch_bounds__(256) void k_proj(
    const float* __restrict__ x,
    const unsigned short* __restrict__ wqkv_sw,
    const float* __restrict__ scale, const float* __restrict__ bias,
    float* __restrict__ qout, float* __restrict__ kraw,
    unsigned short* __restrict__ vbf)
{
    int lane = threadIdx.x & 31;
    int wave = threadIdx.x >> 5;
    int r0   = blockIdx.x * 128 + wave * 16;
    int mrow = r0 + (lane & 15);
    int khalf = (lane >> 4) << 3;

    v8f c[9];
#pragma unroll
    for (int nt = 0; nt < 9; ++nt) c[nt] = 0.0f;

    for (int kc = 0; kc < 8; ++kc) {
        int kb = kc * 32 + khalf;
        Frag16 a;
        const float* xp = x + (size_t)mrow * CIN + kb;
#pragma unroll
        for (int j = 0; j < 4; ++j) {
            a.w[j]     = pk_bf16(xp[2 * j],      xp[2 * j + 1]);
            a.w[4 + j] = pk_bf16(xp[16 + 2 * j], xp[17 + 2 * j]);
        }
#pragma unroll
        for (int nt = 0; nt < 9; ++nt) {
            Frag16 bfr;
            const u32x4* bp = (const u32x4*)(wqkv_sw + (((size_t)(kc * 9 + nt) * 32) + lane) * 16);
            bfr.q[0] = bp[0];
            bfr.q[1] = bp[1];
            c[nt] = __builtin_amdgcn_wmma_f32_16x16x32_bf16(
                        false, a.v, false, bfr.v, (short)0, c[nt], false, false);
        }
    }

    int colbase = lane & 15;
#pragma unroll
    for (int nt = 0; nt < 9; ++nt) {
        int col = nt * 16 + colbase;
        float sc = scale[col];
        float bi = bias[col];
#pragma unroll
        for (int r = 0; r < 8; ++r) {
            int m = r0 + r + khalf;
            float val = c[nt][r] * sc + bi;
            int b = m >> 12;
            int n = m & (NPOS - 1);
            if (col < 64) {
                qout[(size_t)m * 64 + col] = val;
            } else if (col < 80) {
                kraw[((size_t)(b * 16 + (col - 64))) * NPOS + n] = val;
            } else {
                vbf[((size_t)(b * 64 + (col - 80))) * NPOS + n] = f2bf(val);
            }
        }
    }
}

// ---------------------------------------------------------------------------
// K2: softmax over 4096 positions, one block per (b, k-channel) row.
// Each thread owns a contiguous 16-element chunk; packed bf16 b32 stores.
// ---------------------------------------------------------------------------
__global__ __launch_bounds__(256) void k_softmax(
    const float* __restrict__ kraw, unsigned short* __restrict__ ksm)
{
    __shared__ float red[256];
    int t = threadIdx.x;
    const float* row = kraw + (size_t)blockIdx.x * NPOS;

    float r[16];
    float m = -1e30f;
#pragma unroll
    for (int i = 0; i < 16; ++i) {
        r[i] = row[t * 16 + i];
        m = fmaxf(m, r[i]);
    }
    red[t] = m; __syncthreads();
    for (int s = 128; s > 0; s >>= 1) {
        if (t < s) red[t] = fmaxf(red[t], red[t + s]);
        __syncthreads();
    }
    m = red[0];
    __syncthreads();

    float sum = 0.0f;
#pragma unroll
    for (int i = 0; i < 16; ++i) {
        r[i] = __expf(r[i] - m);
        sum += r[i];
    }
    red[t] = sum; __syncthreads();
    for (int s = 128; s > 0; s >>= 1) {
        if (t < s) red[t] += red[t + s];
        __syncthreads();
    }
    float inv = 1.0f / red[0];

    unsigned int* orow = (unsigned int*)(ksm + (size_t)blockIdx.x * NPOS);
#pragma unroll
    for (int j = 0; j < 8; ++j)
        orow[t * 8 + j] = pk_bf16(r[2 * j] * inv, r[2 * j + 1] * inv);
}

// ---------------------------------------------------------------------------
// K3: Lc[b,k,v] = sum_n ksm[b,k,n] * v[b,v,n]   — 16x64 GEMM, K=4096, WMMA.
// 16 blocks (one per batch), 4 waves, one 16x16 v-tile per wave.
// ---------------------------------------------------------------------------
__global__ __launch_bounds__(128) void k_lc(
    const unsigned short* __restrict__ ksm,
    const unsigned short* __restrict__ vbf,
    float* __restrict__ Lc)
{
    int lane = threadIdx.x & 31;
    int wave = threadIdx.x >> 5;
    int b    = blockIdx.x;
    int khalf = (lane >> 4) << 3;

    int kch  = lane & 15;                 // A row (k-channel)
    int vcol = wave * 16 + (lane & 15);   // B column (v-channel)
    const unsigned short* arow = ksm + ((size_t)b * 16 + kch) * NPOS;
    const unsigned short* bcol = vbf + ((size_t)b * 64 + vcol) * NPOS;

    v8f c = 0.0f;
    for (int kc = 0; kc < NPOS; kc += 32) {
        Frag16 a, bf;
        a.q[0]  = *(const u32x4*)(arow + kc + khalf);
        a.q[1]  = *(const u32x4*)(arow + kc + khalf + 16);
        bf.q[0] = *(const u32x4*)(bcol + kc + khalf);
        bf.q[1] = *(const u32x4*)(bcol + kc + khalf + 16);
        c = __builtin_amdgcn_wmma_f32_16x16x32_bf16(
                false, a.v, false, bf.v, (short)0, c, false, false);
    }
#pragma unroll
    for (int r = 0; r < 8; ++r) {
        int k = r + khalf;
        Lc[((size_t)b * 16 + k) * 64 + vcol] = c[r];
    }
}

// ---------------------------------------------------------------------------
// K4: positional conv as WMMA GEMM (taps on the K dimension) + final combine.
// ---------------------------------------------------------------------------
#if __has_builtin(__builtin_amdgcn_global_load_async_to_lds_b32) && \
    __has_builtin(__builtin_amdgcn_global_load_async_to_lds_b128)
#define HAVE_ASYNC_LDS 1
typedef __attribute__((address_space(1))) char   glb_char;
typedef __attribute__((address_space(3))) char   lds_char;
typedef __attribute__((address_space(1))) int    glb_int;
typedef __attribute__((address_space(3))) int    lds_int;
typedef __attribute__((address_space(1))) i32x4v glb_int4;
typedef __attribute__((address_space(3))) i32x4v lds_int4;
#endif

// one K-tile of the im2col conv GEMM; TAIL=true handles the padded last tile.
// Patch pointers per element are hoisted so the 16 N-tiles become ds loads
// with immediate offsets (nt*PSTR*2 fits the DS offset field).
template <bool TAIL>
__device__ __forceinline__ void conv_kstep(
    int kc, int lane, int khalf, int px,
    const unsigned short* __restrict__ sw,
    const unsigned short* __restrict__ pbase,
    v8f c[16])
{
    Frag16 a;
    const u32x4* ap = (const u32x4*)(sw + ((size_t)(kc * 32 + lane)) * 16);
    a.q[0] = ap[0];
    a.q[1] = ap[1];

    const unsigned short* aptr[16];
    unsigned int valid = 0;
#pragma unroll
    for (int e = 0; e < 16; ++e) {
        int koff = (e < 8) ? (khalf + e) : (khalf + 8 + e);
        int t = kc * 32 + koff;
        int offe = 0;
        if (!TAIL || t < TAPS) {
            int dr = t / RT;
            int dc = t - dr * RT;
            offe = dr * PSTR + dc + 1;      // +1: patch col origin shift
            valid |= (1u << e);
        }
        aptr[e] = pbase + px + offe;
    }
#pragma unroll
    for (int nt = 0; nt < 16; ++nt) {
        Frag16 bf;
#pragma unroll
        for (int e = 0; e < 16; ++e) {
            unsigned short vv = aptr[e][nt * PSTR];
            if (TAIL) vv = ((valid >> e) & 1u) ? vv : (unsigned short)0;
            bf.u[e] = vv;
        }
        c[nt] = __builtin_amdgcn_wmma_f32_16x16x32_bf16(
                    false, a.v, false, bf.v, (short)0, c[nt], false, false);
    }
}

__global__ __launch_bounds__(256) void k_conv_combine(
    const unsigned short* __restrict__ wsw,
    const unsigned short* __restrict__ vbf,
    const float* __restrict__ q,
    const float* __restrict__ Lc,
    const float* __restrict__ pos_b,
    float* __restrict__ out)
{
    __shared__ unsigned short s_w[KTC * 32 * 16];    // 17408 B, conv A-frags
    __shared__ unsigned short s_patch[8][38 * PSTR]; // 25536 B, per-wave halo
    __shared__ float s_lp[8][16 * 17 + 4];           //  8832 B, per-wave Lp row
    __shared__ float s_lc[8][16];                    //   512 B

    int lane = threadIdx.x & 31;
    int wave = threadIdx.x >> 5;
    int tX = blockIdx.x, tY = blockIdx.y, b = blockIdx.z;
    int khalf = (lane >> 4) << 3;
    int px = lane & 15;

    // ---- stage conv weight fragments (17408 B = 1088 x b128) ----
#ifdef HAVE_ASYNC_LDS
    {
        glb_char* g = (glb_char*)wsw;
        lds_char* l = (lds_char*)s_w;
        for (int i = threadIdx.x; i < KTC * 32 * 2; i += 256)
            __builtin_amdgcn_global_load_async_to_lds_b128(
                (glb_int4*)(g + (size_t)i * 16),
                (lds_int4*)(l + (size_t)i * 16), 0, 0);
        asm volatile("s_wait_asynccnt 0" ::: "memory");
    }
#else
    {
        const u32x4* src = (const u32x4*)wsw;
        u32x4* dst = (u32x4*)s_w;
        for (int i = threadIdx.x; i < KTC * 32 * 2; i += 256) dst[i] = src[i];
    }
#endif
    __syncthreads();

    int r0g = tY * 16 - RAD;        // patch origin row (may be negative)
    int c0g = tX * 16 - RAD - 1;    // patch origin col, EVEN by construction

    for (int vi = 0; vi < 8; ++vi) {
        int vchan = wave * 8 + vi;
        const unsigned short* vimg = vbf + ((size_t)(b * 64 + vchan)) * NPOS;

        // ---- stage 38 x 40(+pad) halo patch for this wave's v-channel ----
#ifdef HAVE_ASYNC_LDS
        {
            glb_char* g = (glb_char*)vimg;
            lds_char* l = (lds_char*)&s_patch[wave][0];
            for (int idx = lane; idx < 38 * 20; idx += 32) {
                int pr = idx / 20;
                int pc = (idx - pr * 20) * 2;         // even column in patch
                int gr = r0g + pr;
                int gc = c0g + pc;                    // even global column
                if (gr >= 0 && gr < 64 && gc >= 0 && gc < 64) {
                    __builtin_amdgcn_global_load_async_to_lds_b32(
                        (glb_int*)(g + (size_t)(gr * 64 + gc) * 2),
                        (lds_int*)(l + ((size_t)pr * PSTR + pc) * 2), 0, 0);
                } else {
                    *(unsigned int*)&s_patch[wave][pr * PSTR + pc] = 0u;
                }
            }
        }
#else
        for (int idx = lane; idx < 38 * 20; idx += 32) {
            int pr = idx / 20;
            int pc = (idx - pr * 20) * 2;
            int gr = r0g + pr;
            int gc = c0g + pc;
            unsigned int val = 0;
            if (gr >= 0 && gr < 64 && gc >= 0 && gc < 64)
                val = *(const unsigned int*)(vimg + gr * 64 + gc);
            *(unsigned int*)&s_patch[wave][pr * PSTR + pc] = val;
        }
#endif
        if (lane < 16)
            s_lc[wave][lane] = Lc[((size_t)b * 16 + lane) * 64 + vchan] + pos_b[lane];
        if (vi < 7)   // prefetch next v-channel image into cache
            __builtin_prefetch(vimg + NPOS, 0, 1);
#ifdef HAVE_ASYNC_LDS
        asm volatile("s_wait_asynccnt 0" ::: "memory");
#endif
        __syncthreads();

        v8f c[16];
#pragma unroll
        for (int nt = 0; nt < 16; ++nt) c[nt] = 0.0f;

        const unsigned short* pbase = &s_patch[wave][0];
#pragma unroll 1
        for (int kc = 0; kc < KTC - 1; ++kc)
            conv_kstep<false>(kc, lane, khalf, px, s_w, pbase, c);
        conv_kstep<true>(KTC - 1, lane, khalf, px, s_w, pbase, c);

        // ---- combine with q, one spatial row (nt) at a time ----
        for (int nt = 0; nt < 16; ++nt) {
#pragma unroll
            for (int r = 0; r < 8; ++r)
                s_lp[wave][(r + khalf) * 17 + px] = c[nt][r];
            __syncthreads();

            int gy = tY * 16 + nt;
            int gx = tX * 16 + px;
            size_t pos = (size_t)b * NPOS + gy * 64 + gx;
            const float* qp = q + pos * 64;
            int h0 = lane >> 4;
#pragma unroll
            for (int hh = 0; hh < 2; ++hh) {
                int h = h0 + hh * 2;
                float acc = 0.0f;
#pragma unroll
                for (int k = 0; k < 16; ++k)
                    acc += qp[h * 16 + k] * (s_lp[wave][k * 17 + px] + s_lc[wave][k]);
                out[pos * 256 + h * 64 + vchan] = acc;
            }
            __syncthreads();
        }
        __syncthreads();
    }
}

// ---------------------------------------------------------------------------
extern "C" void kernel_launch(void* const* d_in, const int* in_sizes, int n_in,
                              void* d_out, int out_size, void* d_ws, size_t ws_size,
                              hipStream_t stream)
{
    const float* x  = (const float*)d_in[0];
    const float* Wq = (const float*)d_in[1];
    const float* Wk = (const float*)d_in[2];
    const float* Wv = (const float*)d_in[3];
    const float* gq = (const float*)d_in[4];
    const float* bq = (const float*)d_in[5];
    const float* mq = (const float*)d_in[6];
    const float* vq = (const float*)d_in[7];
    const float* gv = (const float*)d_in[8];
    const float* bv = (const float*)d_in[9];
    const float* mv = (const float*)d_in[10];
    const float* vv = (const float*)d_in[11];
    const float* pw = (const float*)d_in[12];
    const float* pb = (const float*)d_in[13];

    char* ws = (char*)d_ws;
    unsigned short* wqkv_sw = (unsigned short*)(ws + WQKV_OFF);
    unsigned short* wsw     = (unsigned short*)(ws + WSW_OFF);
    float*          scale   = (float*)(ws + SCALE_OFF);
    float*          bias    = (float*)(ws + BIAS_OFF);
    float*          qout    = (float*)(ws + Q_OFF);
    float*          kraw    = (float*)(ws + KRAW_OFF);
    unsigned short* ksm     = (unsigned short*)(ws + KSM_OFF);
    unsigned short* vbf     = (unsigned short*)(ws + VBF_OFF);
    float*          lc      = (float*)(ws + LC_OFF);
    float*          out     = (float*)d_out;

    // K0: prep
    {
        int total = 8 * 9 * 32 * 16 + KTC * 32 * 16 + NCOL;
        int blocks = (total + 255) / 256;
        k_prep<<<blocks, 256, 0, stream>>>(Wq, Wk, Wv, gq, bq, mq, vq,
                                           gv, bv, mv, vv, pw,
                                           wqkv_sw, wsw, scale, bias);
    }
    // K1: projection GEMM + BN
    k_proj<<<(NB * NPOS) / 128, 256, 0, stream>>>(x, wqkv_sw, scale, bias,
                                                  qout, kraw, vbf);
    // K2: softmax over positions
    k_softmax<<<NB * 16, 256, 0, stream>>>(kraw, ksm);
    // K3: content lambda
    k_lc<<<NB, 128, 0, stream>>>(ksm, vbf, lc);
    // K4: positional conv + combine
    k_conv_combine<<<dim3(4, 4, NB), 256, 0, stream>>>(wsw, vbf, qout, lc, pb, out);
}
